// ScaledDotProductAttention_73306501808134
// MI455X (gfx1250) — matrix operand order repaired and checked by reading, not compiled, not run on hardware
//
#include <hip/hip_runtime.h>
#include <hip/hip_bf16.h>

// MI455X / gfx1250, wave32. WMMA bf16 path: V_WMMA_F32_16X16X32_BF16.

typedef __attribute__((ext_vector_type(16))) __bf16 v16bf;
typedef __attribute__((ext_vector_type(8)))  float  v8f;

#define HEADS   16
#define DKV     64
#define DMODEL  1024
#define BATCH   2
#define NQTOK   2048
#define NKTOK   2048

static __device__ __forceinline__ v8f wmma_bf16(v16bf a, v16bf b, v8f c) {
  // (neg_a, A, neg_b, B, c_mod, C, reuse_a, reuse_b)
  return __builtin_amdgcn_wmma_f32_16x16x32_bf16(false, a, false, b, (short)0, c,
                                                 false, false);
}

// ---------------------------------------------------------------------------
// Projection GEMM: out = X[4096,1024] @ W[1024,1024] + b, written as bf16.
// Block tile: 16 rows x 128 cols (8 waves, one 16x16 tile each, shared M-tile).
// X (16x32) and W (32x128) tiles are staged f32->bf16 in LDS with coalesced
// b128/b64 global loads; WMMA fragments are built from LDS.
// VT==0 : out[b,h,n,d]   (Q, K layout — contiguous d for score B-frags)
// VT==1 : out[b,h,d,n]   (V layout — contiguous n for PV B-frags)
// ---------------------------------------------------------------------------
template <int VT>
__global__ __launch_bounds__(256) void proj_kernel(
    const float* __restrict__ X, const float* __restrict__ W,
    const float* __restrict__ bias, __bf16* __restrict__ out)
{
  __shared__ __bf16 Xs[16][32];     // 1 KB
  __shared__ __bf16 Ws[32][128];    // 8 KB

  const int tid   = threadIdx.x;
  const int lane  = tid & 31;
  const int wid   = tid >> 5;
  const int mt    = blockIdx.x >> 3;           // 256 row tiles
  const int cb    = (blockIdx.x & 7) * 128;    // 8 col blocks of 128
  const int m0    = mt * 16;
  const int col16 = lane & 15;
  const int half  = lane >> 4;                 // 0: lanes 0-15, 1: lanes 16-31
  const int akb   = half ? 16 : 0;             // A-matrix K base (contiguous, matches B)
  const int bkb   = half ? 16 : 0;             // B-matrix K base
  const int c     = cb + wid * 16 + col16;     // this lane's output column

  // cooperative-load index precompute
  const int xr = tid >> 4, xc = (tid & 15) * 2;          // X: 2 f32 / thread
  const int wr = tid >> 3, wc = (tid & 7) * 16;          // W: 16 f32 / thread

  v8f acc = {};
  for (int kk = 0; kk < DMODEL; kk += 32) {
    // ---- stage X tile [16 x 32] f32 -> bf16, coalesced float2 loads ----
    {
      const float2 x2 = *(const float2*)(X + (size_t)(m0 + xr) * DMODEL + kk + xc);
      Xs[xr][xc]     = (__bf16)x2.x;
      Xs[xr][xc + 1] = (__bf16)x2.y;
    }
    // ---- stage W tile [32 x 128] f32 -> bf16, coalesced float4 loads ----
    {
      const float* wrow = W + (size_t)(kk + wr) * (HEADS * DKV) + cb + wc;
#pragma unroll
      for (int j = 0; j < 4; ++j) {
        const float4 w4 = *(const float4*)(wrow + 4 * j);
        Ws[wr][wc + 4 * j + 0] = (__bf16)w4.x;
        Ws[wr][wc + 4 * j + 1] = (__bf16)w4.y;
        Ws[wr][wc + 4 * j + 2] = (__bf16)w4.z;
        Ws[wr][wc + 4 * j + 3] = (__bf16)w4.w;
      }
    }
    __syncthreads();

    // ---- fragments from LDS ----
    v16bf a;
#pragma unroll
    for (int e = 0; e < 8; ++e) a[e]     = Xs[col16][akb + e];
#pragma unroll
    for (int e = 0; e < 8; ++e) a[8 + e] = Xs[col16][akb + 8 + e];
    v16bf bf;
#pragma unroll
    for (int e = 0; e < 16; ++e) bf[e] = Ws[bkb + e][wid * 16 + col16];

    acc = wmma_bf16(a, bf, acc);
    __syncthreads();                // fragments consumed before next overwrite
  }

  const float bval = bias[c];
  const int h = c >> 6, d = c & 63;
#pragma unroll
  for (int r = 0; r < 8; ++r) {
    const int m = m0 + (half ? r + 8 : r);    // C-layout row mapping
    const int b = m >> 11, n = m & 2047;
    const float val = acc[r] + bval;
    if (VT)
      out[((size_t)(b * HEADS + h) * DKV + d) * NKTOK + n] = (__bf16)val;
    else
      out[((size_t)(b * HEADS + h) * NKTOK + n) * DKV + d] = (__bf16)val;
  }
}

// ---------------------------------------------------------------------------
// Flash attention with post-softmax group_prob gate folded into PV numerator:
//   out = (sum_k e^{s_k - m} * gp_k * v_k) / (sum_k e^{s_k - m})
// One wave handles a 16-row Q tile for one (b,h); streams 32 keys per step.
// ---------------------------------------------------------------------------
__global__ __launch_bounds__(256) void attn_kernel(
    const __bf16* __restrict__ qw, const __bf16* __restrict__ kw,
    const __bf16* __restrict__ vt, const float* __restrict__ gp,
    float* __restrict__ out)
{
  __shared__ __bf16 lds_p[8][16][32];         // per-wave P-tile staging (1 KB)

  const int lane  = threadIdx.x & 31;
  const int wid   = threadIdx.x >> 5;
  const int h     = blockIdx.y;
  const int b     = blockIdx.z;
  const int q0    = blockIdx.x * 128 + wid * 16;
  const int col16 = lane & 15;
  const int half  = lane >> 4;
  const int akb   = half ? 16 : 0;
  const size_t bh = (size_t)(b * HEADS + h);

  // Q tile as two 16x32 A fragments (d = 0..31, 32..63), loaded once.
  v16bf qa[2];
  {
    const __bf16* qrow = qw + (bh * NQTOK + q0 + col16) * DKV;
#pragma unroll
    for (int cc = 0; cc < 2; ++cc) {
#pragma unroll
      for (int e = 0; e < 8; ++e) qa[cc][e]     = qrow[cc * 32 + akb + e];
#pragma unroll
      for (int e = 0; e < 8; ++e) qa[cc][8 + e] = qrow[cc * 32 + akb + 8 + e];
    }
  }

  v8f acc[4] = {};                            // output 16x64 (4 d-tiles), f32
  float mrow[8], lrow[8];
#pragma unroll
  for (int r = 0; r < 8; ++r) { mrow[r] = -__builtin_inff(); lrow[r] = 0.f; }

  const float* gpb = gp + (size_t)b * NQTOK * NKTOK;

  for (int k0 = 0; k0 < NKTOK; k0 += 32) {
    // --- scores: two 16x16 tiles, K-dim = 64 -> 2 WMMAs each ---
    float sv[2][8];
#pragma unroll
    for (int t = 0; t < 2; ++t) {
      const int key = k0 + t * 16 + col16;
      const __bf16* krow = kw + (bh * NKTOK + key) * DKV;
      __builtin_prefetch(krow + 32 * DKV, 0, 1);   // next k-step, same lane
      v8f s = {};
#pragma unroll
      for (int cc = 0; cc < 2; ++cc) {
        const int dbase = cc * 32 + (half ? 16 : 0);
        v16bf kb = *(const v16bf*)(krow + dbase);  // 32B contiguous per lane
        s = wmma_bf16(qa[cc], kb, s);
      }
#pragma unroll
      for (int r = 0; r < 8; ++r) sv[t][r] = s[r] * 0.125f;  // 1/sqrt(64)
    }

    // --- online softmax update (rows striped over 16-lane halves) ---
#pragma unroll
    for (int r = 0; r < 8; ++r) {
      float pm = fmaxf(sv[0][r], sv[1][r]);
#pragma unroll
      for (int msk = 1; msk <= 8; msk <<= 1)
        pm = fmaxf(pm, __shfl_xor(pm, msk, 32));   // reduce within half-wave
      const float mnew    = fmaxf(mrow[r], pm);
      const float rescale = __expf(mrow[r] - mnew);
      mrow[r] = mnew;
      lrow[r] *= rescale;

      const int row = half ? r + 8 : r;
      float lsum = 0.f;
#pragma unroll
      for (int t = 0; t < 2; ++t) {
        const float p = __expf(sv[t][r] - mnew);
        lsum += p;                                  // denominator: no gate
        const float g = gpb[(size_t)(q0 + row) * NKTOK + (k0 + t * 16 + col16)];
        lds_p[wid][row][t * 16 + col16] = (__bf16)(p * g);  // gated numerator
      }
#pragma unroll
      for (int msk = 1; msk <= 8; msk <<= 1)
        lsum += __shfl_xor(lsum, msk, 32);
      lrow[r] += lsum;
#pragma unroll
      for (int dt = 0; dt < 4; ++dt) acc[dt][r] *= rescale;
    }

    // wave-local LDS transpose: C-layout P -> A-layout fragment
    asm volatile("s_wait_dscnt 0" ::: "memory");
    v16bf pa;
    {
      const __bf16* prow = &lds_p[wid][col16][0];
#pragma unroll
      for (int e = 0; e < 8; ++e) pa[e]     = prow[akb + e];
#pragma unroll
      for (int e = 0; e < 8; ++e) pa[8 + e] = prow[akb + 8 + e];
    }

    // --- PV: 4 d-tiles, V^T layout gives contiguous 32B B-fragments ---
#pragma unroll
    for (int dt = 0; dt < 4; ++dt) {
      const int d = dt * 16 + col16;
      const __bf16* vrow = vt + (bh * DKV + d) * NKTOK + k0 + (half ? 16 : 0);
      __builtin_prefetch(vrow + 32, 0, 1);
      v16bf vb = *(const v16bf*)vrow;
      acc[dt] = wmma_bf16(pa, vb, acc[dt]);
    }
    asm volatile("s_wait_dscnt 0" ::: "memory");   // reads done before rewrite
  }

  // --- normalize and store [b, q, h*64 + d] as f32 ---
#pragma unroll
  for (int r = 0; r < 8; ++r) {
    const int row = half ? r + 8 : r;
    const float inv = 1.0f / lrow[r];
    const size_t obase = ((size_t)b * NQTOK + q0 + row) * DMODEL + h * DKV;
#pragma unroll
    for (int dt = 0; dt < 4; ++dt)
      out[obase + dt * 16 + col16] = acc[dt][r] * inv;
  }
}

// ---------------------------------------------------------------------------
extern "C" void kernel_launch(void* const* d_in, const int* in_sizes, int n_in,
                              void* d_out, int out_size, void* d_ws, size_t ws_size,
                              hipStream_t stream) {
  (void)in_sizes; (void)n_in; (void)out_size; (void)ws_size;
  const float* queries = (const float*)d_in[0];
  const float* keys    = (const float*)d_in[1];
  const float* values  = (const float*)d_in[2];
  const float* gp      = (const float*)d_in[3];
  // d_in[4] attention_mask: reference discards masked_fill result -> unused.
  const float* Wq = (const float*)d_in[5];
  const float* bq = (const float*)d_in[6];
  const float* Wk = (const float*)d_in[7];
  const float* bk = (const float*)d_in[8];
  const float* Wv = (const float*)d_in[9];
  const float* bv = (const float*)d_in[10];
  float* out = (float*)d_out;

  // Workspace: 3 bf16 projection tensors, 8 MB each (24 MB total).
  const size_t proj_elems = (size_t)BATCH * HEADS * NKTOK * DKV;
  __bf16* qw = (__bf16*)d_ws;
  __bf16* kw = (__bf16*)((char*)d_ws + proj_elems * sizeof(__bf16));
  __bf16* vt = (__bf16*)((char*)d_ws + 2 * proj_elems * sizeof(__bf16));

  // 2048 blocks: 256 M-tiles x 8 col-blocks (128 cols each), 8 waves/block.
  dim3 pgrid(2048), pblk(256);
  proj_kernel<0><<<pgrid, pblk, 0, stream>>>(queries, Wq, bq, qw);
  proj_kernel<0><<<pgrid, pblk, 0, stream>>>(keys,    Wk, bk, kw);
  proj_kernel<1><<<pgrid, pblk, 0, stream>>>(values,  Wv, bv, vt);

  dim3 agrid(NQTOK / 128, HEADS, BATCH), ablk(256);
  attn_kernel<<<agrid, ablk, 0, stream>>>(qw, kw, vt, gp, out);
}
